// SimpleGNN3_47699906789972
// MI455X (gfx1250) — matrix operand order repaired and checked by reading, not compiled, not run on hardware
//
#include <hip/hip_runtime.h>

#define NB 4
#define NSZ 128
#define FCH 32
#define DCH 64
#define NN (NSZ*NSZ)
#define PD ((size_t)NB*DCH*NN)

typedef __bf16 bf;
typedef __attribute__((ext_vector_type(16))) __bf16 v16bf;
typedef __attribute__((ext_vector_type(8)))  float  v8f;

// ---------------- WMMA helpers (wave32, 16x16x32 bf16 -> f32) ----------------

__device__ __forceinline__ v8f wmma_bf16(v16bf a, v16bf b, v8f c) {
  return __builtin_amdgcn_wmma_f32_16x16x32_bf16(false, a, false, b, (short)0, c,
                                                 false, false);
}

// A-operand (16x32 bf16) from row-major tile; lane L<16: row=L, K=0..7,16..23;
// lane L>=16: row=L-16, K=8..15,24..31  (per ISA 7.12.2 layout)
__device__ __forceinline__ v16bf frag_a_rows(const bf* tile, int rs, int lane) {
  int row = lane & 15, kb = (lane >> 4) * 8;
  const bf* p = tile + (size_t)row * rs + kb;
  union { v16bf v; uint4 q[2]; } u;
  u.q[0] = *reinterpret_cast<const uint4*>(p);
  u.q[1] = *reinterpret_cast<const uint4*>(p + 16);
  return u.v;
}

// B-operand (32x16 bf16) from *transposed* right matrix stored row-major [col][k]
// lane: col = lane&15, K = (lane>>4)*8 + 0..7, 16 + (lane>>4)*8 + 0..7
__device__ __forceinline__ v16bf frag_b_rows(const bf* rt, int rs, int lane) {
  int col = lane & 15, kb = (lane >> 4) * 8;
  const bf* p = rt + (size_t)col * rs + kb;
  union { v16bf v; uint4 q[2]; } u;
  u.q[0] = *reinterpret_cast<const uint4*>(p);
  u.q[1] = *reinterpret_cast<const uint4*>(p + 16);
  return u.v;
}

// A-operand gathered along the d dimension (stride NN elements)
__device__ __forceinline__ v16bf gather_dfrag(const bf* T, int d0, int kb8) {
  v16bf a;
#pragma unroll
  for (int e = 0; e < 8; ++e) a[e] = T[(size_t)(d0 + 2 * kb8 + e) * NN];
#pragma unroll
  for (int e = 0; e < 8; ++e) a[8 + e] = T[(size_t)(d0 + 2 * kb8 + 8 + e) * NN];
  return a;
}

// 32-byte fragment load from LDS (per-lane v16bf image)
__device__ __forceinline__ v16bf ld_frag_lds(const bf* p) {
  union { v16bf v; uint4 q[2]; } u;
  const uint4* s = reinterpret_cast<const uint4*>(p);
  u.q[0] = s[0];
  u.q[1] = s[1];
  return u.v;
}

// ---------------- kernels ----------------

__global__ void k_zero(float* p, int count) {
  int i = blockIdx.x * 256 + threadIdx.x;
  if (i < count) p[i] = 0.0f;
}

__global__ void k_scatter(const int* ei, const float* ew, const int* batch,
                          float* A, int E) {
  int e = blockIdx.x * 256 + threadIdx.x;
  if (e >= E) return;
  int s = ei[e], t = ei[E + e];
  int g = batch[s];
  int r = s - g * NSZ, c = t - g * NSZ;
  atomicAdd(&A[((size_t)g * NSZ + r) * NSZ + c], ew[e]);
}

// per (b,i): rowsum(A), bf16 copy, bf16 transpose
__global__ void k_adj_fin(const float* A, bf* Abf, bf* ATbf, float* rA) {
  int bi = blockIdx.x;            // b*NSZ + i
  int b = bi >> 7;
  int i = bi & 127;
  int j = threadIdx.x;            // 128 threads
  float a = A[(size_t)bi * NSZ + j];
  Abf[(size_t)bi * NSZ + j] = (bf)a;
  ATbf[(size_t)b * NN + (size_t)j * NSZ + i] = (bf)a;
  float s = a;
#pragma unroll
  for (int o = 16; o; o >>= 1) s += __shfl_down(s, o);
  __shared__ float ls[4];
  if ((j & 31) == 0) ls[j >> 5] = s;
  __syncthreads();
  if (j == 0) rA[bi] = ls[0] + ls[1] + ls[2] + ls[3];
}

// B3 = A @ A^T per graph, one wave per 16x16 tile (WMMA)
__global__ void k_b3(const bf* Abf, float* B3) {
  int t = blockIdx.x;
  int jt = t & 7, it = (t >> 3) & 7, b = t >> 6;
  int lane = threadIdx.x;
  const bf* base = Abf + (size_t)b * NN;
  v8f acc = {};
#pragma unroll
  for (int kc = 0; kc < NSZ; kc += 32) {
    v16bf af = frag_a_rows(base + (size_t)(it * 16) * NSZ + kc, NSZ, lane);
    // right matrix is A^T, so transposed-right storage == A itself
    v16bf bg = frag_b_rows(base + (size_t)(jt * 16) * NSZ + kc, NSZ, lane);
    acc = wmma_bf16(af, bg, acc);
  }
  int col = lane & 15, ro = (lane >> 4) * 8;
#pragma unroll
  for (int r = 0; r < 8; ++r)
    B3[((size_t)b * NSZ + it * 16 + ro + r) * NSZ + jt * 16 + col] = acc[r];
}

// x [b*n, f] -> X0 [b][2f][n][n]
__global__ void k_build_x2d(const float* x, float* X0) {
  size_t idx = (size_t)blockIdx.x * 256 + threadIdx.x;
  int j = (int)(idx & 127);
  int i = (int)((idx >> 7) & 127);
  int c = (int)((idx >> 14) & 63);
  int b = (int)(idx >> 20);
  float v = (c < FCH) ? x[((size_t)(b * NSZ + i)) * FCH + c]
                      : x[((size_t)(b * NSZ + j)) * FCH + (c - FCH)];
  X0[idx] = v;
}

// per (b,d): W = X*A, bf16 copies + transposes, row/col sums of W and X
__global__ void k_prep(const float* Xf, const float* A, bf* Xb, bf* XTb, bf* Wb,
                       bf* WTb, float* v4, float* v5, float* srow, float* scol) {
  int blk = blockIdx.x;           // b*DCH + d
  int b = blk >> 6;
  int j = threadIdx.x;            // 128 threads, one column each
  const float* Ab = A + (size_t)b * NN;
  size_t basep = (size_t)blk * NN;
  float accv5 = 0.0f, accsc = 0.0f;
  __shared__ float ls[8];
  for (int i = 0; i < NSZ; ++i) {
    float xv = Xf[basep + (size_t)i * NSZ + j];
    float av = Ab[(size_t)i * NSZ + j];
    float wv = xv * av;
    Xb[basep + (size_t)i * NSZ + j] = (bf)xv;
    Wb[basep + (size_t)i * NSZ + j] = (bf)wv;
    XTb[basep + (size_t)j * NSZ + i] = (bf)xv;
    WTb[basep + (size_t)j * NSZ + i] = (bf)wv;
    accv5 += wv; accsc += xv;
    float sw = wv, sx = xv;
#pragma unroll
    for (int o = 16; o; o >>= 1) { sw += __shfl_down(sw, o); sx += __shfl_down(sx, o); }
    if ((j & 31) == 0) { ls[j >> 5] = sw; ls[4 + (j >> 5)] = sx; }
    __syncthreads();
    if (j == 0) {
      v4[(size_t)blk * NSZ + i] = ls[0] + ls[1] + ls[2] + ls[3];
      srow[(size_t)blk * NSZ + i] = ls[4] + ls[5] + ls[6] + ls[7];
    }
    __syncthreads();
  }
  v5[(size_t)blk * NSZ + j] = accv5;
  scol[(size_t)blk * NSZ + j] = accsc;
}

// per (b,d) tile: M1 = W^T A, M2 = W A, M6 = X^T A (+ transposed copies), WMMA
__global__ void k_gemm_bd(const bf* WTb, const bf* Wb, const bf* XTb,
                          const bf* ATbf, bf* M1, bf* M1T, bf* M2, bf* M2T,
                          bf* M6, bf* M6T) {
  int t = blockIdx.x;
  int jt = t & 7, it = (t >> 3) & 7;
  int bd = t >> 6;
  int b = bd >> 6;
  int lane = threadIdx.x;
  size_t mb = (size_t)bd * NN;
  const bf* bt = ATbf + (size_t)b * NN + (size_t)(jt * 16) * NSZ;
  v8f a1 = {}, a2 = {}, a6 = {};
#pragma unroll
  for (int kc = 0; kc < NSZ; kc += 32) {
    v16bf bg = frag_b_rows(bt + kc, NSZ, lane);
    v16bf f1 = frag_a_rows(WTb + mb + (size_t)(it * 16) * NSZ + kc, NSZ, lane);
    a1 = wmma_bf16(f1, bg, a1);
    v16bf f2 = frag_a_rows(Wb + mb + (size_t)(it * 16) * NSZ + kc, NSZ, lane);
    a2 = wmma_bf16(f2, bg, a2);
    v16bf f6 = frag_a_rows(XTb + mb + (size_t)(it * 16) * NSZ + kc, NSZ, lane);
    a6 = wmma_bf16(f6, bg, a6);
  }
  int col = lane & 15, ro = (lane >> 4) * 8;
#pragma unroll
  for (int r = 0; r < 8; ++r) {
    int rr = it * 16 + ro + r, cc = jt * 16 + col;
    size_t pij = mb + (size_t)rr * NSZ + cc;
    size_t pji = mb + (size_t)cc * NSZ + rr;
    M1[pij] = (bf)a1[r]; M1T[pji] = (bf)a1[r];
    M2[pij] = (bf)a2[r]; M2T[pji] = (bf)a2[r];
    M6[pij] = (bf)a6[r]; M6T[pji] = (bf)a6[r];
  }
}

// fold coeffs into contraction B-operands: G[c][e][d] bf16, c = 10 channels
__global__ void k_coeffG(const float* cf, bf* G, float norm) {
  int t = blockIdx.x * 256 + threadIdx.x;
  if (t >= DCH * DCH) return;
  int d = t >> 6, e = t & 63;
  const float* c = cf + ((size_t)d * DCH + e) * 20;
  float g[10] = { c[0], c[1], c[2], c[3],
                  c[9] + c[11], c[10] + c[12],
                  c[4], c[13], c[14], c[19] * norm };
#pragma unroll
  for (int k = 0; k < 10; ++k)
    G[(size_t)k * (DCH * DCH) + (size_t)e * DCH + d] = (bf)g[k];
}

// row/col broadcast vector contributions
__global__ void k_vecs(const float* cf, const float* v4, const float* v5,
                       const float* srow, const float* scol, float* rowv,
                       float* colv) {
  int t = blockIdx.x * 256 + threadIdx.x;   // b*64*NSZ threads
  int i = t & 127;
  int e = (t >> 7) & 63;
  int b = t >> 13;
  size_t vb = (size_t)b * DCH * NSZ;
  float rv = 0.0f, cv = 0.0f;
  for (int d = 0; d < DCH; ++d) {
    const float* c = cf + ((size_t)d * DCH + e) * 20;
    size_t o = vb + (size_t)d * NSZ + i;
    float a = v4[o], w = v5[o], s = srow[o], q = scol[o];
    rv += c[5] * a + c[7] * w + c[15] * s + c[18] * q;
    cv += c[6] * a + c[8] * w + c[16] * s + c[17] * q;
  }
  rowv[((size_t)(b * DCH + e)) * NSZ + i] = rv;
  colv[((size_t)(b * DCH + e)) * NSZ + i] = cv;
}

// main contraction.
// Block = 4 waves = one (b, i, j-tile) position tile; wave w owns e-tile w.
// The 14 A-operand fragments (12 matrix-channel chunks + 2 X chunks) are
// gathered from global ONCE, cooperatively, into LDS (stored in per-lane
// v16bf image), then all 4 waves consume them via ds_load_b128.
__global__ void k_contract(const bf* M1, const bf* M1T, const bf* M2,
                           const bf* M2T, const bf* M6, const bf* M6T,
                           const bf* Xb, const bf* G, const float* B3,
                           const float* rA, const float* rowv,
                           const float* colv, const float* bias, float* Y,
                           int do_sig) {
  int t = blockIdx.x;
  int jt = t & 7;
  int i  = (t >> 3) & 127;
  int b  = t >> 10;
  int wave = threadIdx.x >> 5;    // e-tile index (0..3)
  int lane = threadIdx.x & 31;
  int row = lane & 15, kb8 = (lane >> 4) * 8, kb16 = (lane >> 4) * 16;
  int jbase = jt * 16;
  const bf* Ts[6] = { M1, M1T, M2, M2T, M6, M6T };
  size_t posoff = (size_t)(b * DCH) * NN + (size_t)i * NSZ + jbase + row;

  __shared__ bf smem[14][32][16];   // 14 fragments x 32 lanes x 16 bf16 = 14 KB

  // --- cooperative gather: wave w fills fragments w, w+4, w+8, (w+12) ---
  for (int q = wave; q < 14; q += 4) {
    const bf* T = (q < 12) ? Ts[q >> 1] + posoff : Xb + posoff;
    int d0 = (q < 12) ? (q & 1) * 32 : (q - 12) * 32;
    v16bf av = gather_dfrag(T, d0, kb8);
    union { v16bf v; uint4 q4[2]; } u;
    u.v = av;
    uint4* dst = reinterpret_cast<uint4*>(&smem[q][lane][0]);
    dst[0] = u.q4[0];
    dst[1] = u.q4[1];
  }
  __syncthreads();

  // --- consume: 20 WMMAs per wave against this e-tile's B-operands ---
  int et = wave;
  v8f acc = {}, s6 = {}, s7 = {}, s8 = {}, s9 = {};
  const bf* gb = G + (size_t)(et * 16 + (lane & 15)) * DCH + kb16;
#pragma unroll
  for (int q = 0; q < 12; ++q) {
    int c = q >> 1, d0 = (q & 1) * 32;
    v16bf av = ld_frag_lds(&smem[q][lane][0]);
    const bf* g = gb + (size_t)c * (DCH * DCH);
    union { v16bf v; uint4 q4[2]; } ug;
    ug.q4[0] = *reinterpret_cast<const uint4*>(g + d0);
    ug.q4[1] = *reinterpret_cast<const uint4*>(g + d0 + 8);
    acc = wmma_bf16(av, ug.v, acc);
  }
#pragma unroll
  for (int q = 12; q < 14; ++q) {
    int d0 = (q - 12) * 32;
    v16bf av = ld_frag_lds(&smem[q][lane][0]);
    union { v16bf v; uint4 q4[2]; } u6, u7, u8, u9;
    const bf* g6 = gb + (size_t)6 * (DCH * DCH);
    const bf* g7 = gb + (size_t)7 * (DCH * DCH);
    const bf* g8 = gb + (size_t)8 * (DCH * DCH);
    const bf* g9 = gb + (size_t)9 * (DCH * DCH);
    u6.q4[0] = *reinterpret_cast<const uint4*>(g6 + d0);
    u6.q4[1] = *reinterpret_cast<const uint4*>(g6 + d0 + 8);
    u7.q4[0] = *reinterpret_cast<const uint4*>(g7 + d0);
    u7.q4[1] = *reinterpret_cast<const uint4*>(g7 + d0 + 8);
    u8.q4[0] = *reinterpret_cast<const uint4*>(g8 + d0);
    u8.q4[1] = *reinterpret_cast<const uint4*>(g8 + d0 + 8);
    u9.q4[0] = *reinterpret_cast<const uint4*>(g9 + d0);
    u9.q4[1] = *reinterpret_cast<const uint4*>(g9 + d0 + 8);
    s6 = wmma_bf16(av, u6.v, s6);
    s7 = wmma_bf16(av, u7.v, s7);
    s8 = wmma_bf16(av, u8.v, s8);
    s9 = wmma_bf16(av, u9.v, s9);
  }

  // --- epilogue: position scalars, broadcast vectors, bias, sigmoid ---
  int e = et * 16 + (lane & 15);
  int ro = (lane >> 4) * 8;
  float rAi = rA[b * NSZ + i];
  float rv = rowv[((size_t)(b * DCH + e)) * NSZ + i];
  float bs = bias[e];
#pragma unroll
  for (int r = 0; r < 8; ++r) {
    int j = jbase + ro + r;
    float val = acc[r]
              + B3[((size_t)b * NSZ + i) * NSZ + j] * s6[r]
              + rAi * s7[r]
              + rA[b * NSZ + j] * s8[r]
              + s9[r]
              + rv
              + colv[((size_t)(b * DCH + e)) * NSZ + j];
    val = val * (1.0f / (float)NSZ) + bs;
    if (do_sig) val = 1.0f / (1.0f + __expf(-val));
    Y[((size_t)(b * DCH + e) * NSZ + i) * NSZ + j] = val;
  }
}

__global__ void k_reduce(const float* Y, float* out) {
  int b = blockIdx.x;
  const size_t tot = (size_t)DCH * NN;
  const float* p = Y + (size_t)b * tot;
  float s = 0.0f;
  for (size_t idx = threadIdx.x; idx < tot; idx += 256) s += p[idx];
#pragma unroll
  for (int o = 16; o; o >>= 1) s += __shfl_down(s, o);
  __shared__ float ls[8];
  if ((threadIdx.x & 31) == 0) ls[threadIdx.x >> 5] = s;
  __syncthreads();
  if (threadIdx.x == 0) {
    float tsum = 0.0f;
#pragma unroll
    for (int k = 0; k < 8; ++k) tsum += ls[k];
    out[b] = tsum / (float)tot;
  }
}

// ---------------- host launch ----------------

extern "C" void kernel_launch(void* const* d_in, const int* in_sizes, int n_in,
                              void* d_out, int out_size, void* d_ws,
                              size_t ws_size, hipStream_t stream) {
  const float* x       = (const float*)d_in[0];
  const float* ew      = (const float*)d_in[1];
  const float* coeffs1 = (const float*)d_in[2];
  const float* bias1   = (const float*)d_in[3];
  const float* coeffs2 = (const float*)d_in[4];
  const float* bias2   = (const float*)d_in[5];
  const int*   ei      = (const int*)d_in[6];
  const int*   batch   = (const int*)d_in[7];
  int E = in_sizes[1];

  // workspace carve-up (256B aligned)
  char* w = (char*)d_ws;
  auto carve = [&](size_t bytes) -> void* {
    void* p = (void*)w;
    w += (bytes + 255) & ~(size_t)255;
    return p;
  };
  float* A    = (float*)carve((size_t)NB * NN * 4);
  float* B3   = (float*)carve((size_t)NB * NN * 4);
  float* rA   = (float*)carve((size_t)NB * NSZ * 4);
  float* v4   = (float*)carve((size_t)NB * DCH * NSZ * 4);
  float* v5   = (float*)carve((size_t)NB * DCH * NSZ * 4);
  float* srow = (float*)carve((size_t)NB * DCH * NSZ * 4);
  float* scol = (float*)carve((size_t)NB * DCH * NSZ * 4);
  float* rowv = (float*)carve((size_t)NB * DCH * NSZ * 4);
  float* colv = (float*)carve((size_t)NB * DCH * NSZ * 4);
  float* X0   = (float*)carve(PD * 4);
  float* X1   = (float*)carve(PD * 4);
  bf* Abf  = (bf*)carve((size_t)NB * NN * 2);
  bf* ATbf = (bf*)carve((size_t)NB * NN * 2);
  bf* Xb   = (bf*)carve(PD * 2);
  bf* XTb  = (bf*)carve(PD * 2);
  bf* Wb   = (bf*)carve(PD * 2);
  bf* WTb  = (bf*)carve(PD * 2);
  bf* M1   = (bf*)carve(PD * 2);
  bf* M1T  = (bf*)carve(PD * 2);
  bf* M2   = (bf*)carve(PD * 2);
  bf* M2T  = (bf*)carve(PD * 2);
  bf* M6   = (bf*)carve(PD * 2);
  bf* M6T  = (bf*)carve(PD * 2);
  bf* G    = (bf*)carve((size_t)10 * DCH * DCH * 2);

  // adjacency
  k_zero<<<(NB * NN + 255) / 256, 256, 0, stream>>>(A, NB * NN);
  k_scatter<<<(E + 255) / 256, 256, 0, stream>>>(ei, ew, batch, A, E);
  k_adj_fin<<<NB * NSZ, 128, 0, stream>>>(A, Abf, ATbf, rA);
  k_b3<<<NB * 64, 32, 0, stream>>>(Abf, B3);

  // x_2d (layer-0 input)
  k_build_x2d<<<(unsigned)(PD / 256), 256, 0, stream>>>(x, X0);

  for (int layer = 0; layer < 2; ++layer) {
    const float* Xf = (layer == 0) ? X0 : X1;
    float* Y        = (layer == 0) ? X1 : X0;   // reuse buffers
    const float* cf = (layer == 0) ? coeffs1 : coeffs2;
    const float* bi = (layer == 0) ? bias1 : bias2;
    int sig = (layer == 0) ? 1 : 0;

    k_prep<<<NB * DCH, 128, 0, stream>>>(Xf, A, Xb, XTb, Wb, WTb, v4, v5, srow,
                                         scol);
    k_gemm_bd<<<NB * DCH * 64, 32, 0, stream>>>(WTb, Wb, XTb, ATbf, M1, M1T,
                                                M2, M2T, M6, M6T);
    k_coeffG<<<(DCH * DCH + 255) / 256, 256, 0, stream>>>(cf, G, (float)NSZ);
    k_vecs<<<(NB * DCH * NSZ + 255) / 256, 256, 0, stream>>>(cf, v4, v5, srow,
                                                             scol, rowv, colv);
    k_contract<<<NB * NSZ * 8, 128, 0, stream>>>(M1, M1T, M2, M2T, M6, M6T,
                                                 Xb, G, B3, rA, rowv, colv,
                                                 bi, Y, sig);
  }

  k_reduce<<<NB, 256, 0, stream>>>(X0, (float*)d_out);
}
